// Encoder_46866683134741
// MI455X (gfx1250) — matrix operand order repaired and checked by reading, not compiled, not run on hardware
//
#include <hip/hip_runtime.h>

// ---------------- problem constants ----------------
#define LNUM 2
#define BB   2
#define SS   2048
#define DD   1024
#define HH   16
#define DHH  64
#define FF   4096
#define BSS  (BB*SS)   // 4096 tokens

typedef unsigned short u16;
typedef unsigned int   u32;
typedef unsigned long long u64;
typedef __attribute__((ext_vector_type(16))) __bf16 v16bf;
typedef __attribute__((ext_vector_type(8)))  float  v8f;
typedef __attribute__((ext_vector_type(4)))  u32    uint4v;
typedef __attribute__((ext_vector_type(4)))  float  float4v;
typedef __attribute__((ext_vector_type(4)))  u16    ushort4v;

union FragU { uint4v q[2]; v16bf v; };
static_assert(sizeof(FragU) == 32, "frag size");

__device__ __forceinline__ u16 f2bf(float f) {
    u32 u = __float_as_uint(f);
    u32 r = (u + 0x7FFFu + ((u >> 16) & 1u)) >> 16;   // round-to-nearest-even
    return (u16)r;
}

// CDNA5 async global->LDS copy, 16B per lane (GV mode), tracked by ASYNCcnt.
__device__ __forceinline__ void async_b128(unsigned lds, u64 gaddr) {
    asm volatile("global_load_async_to_lds_b128 %0, %1, off"
                 :: "v"(lds), "v"(gaddr) : "memory");
}
__device__ __forceinline__ void wait_async0() {
    asm volatile("s_wait_asynccnt 0x0" ::: "memory");
}
__device__ __forceinline__ unsigned lds_addr(const void* p) {
    return (unsigned)(size_t)p;   // low 32 bits of flat shared address = LDS byte offset
}

// ---------------- fp32 -> bf16 convert ----------------
__global__ __launch_bounds__(256) void conv_bf16_kernel(const float* __restrict__ in,
                                                        u16* __restrict__ out, int n4) {
    int i = blockIdx.x * 256 + threadIdx.x;
    if (i < n4) {
        float4v f = ((const float4v*)in)[i];
        ushort4v o;
        o.x = f2bf(f.x); o.y = f2bf(f.y); o.z = f2bf(f.z); o.w = f2bf(f.w);
        ((ushort4v*)out)[i] = o;
    }
}

// ---------------- generic bf16 GEMM: C[M,N] = A[M,K]@B[K,N] (+bias, relu) ----------------
// block tile 128x128, k-step 32, 256 threads = 8 waves (4x2), wave tile 32x64.
// Double-buffered LDS filled by async global->LDS loads (ASYNCcnt pipeline).
template<bool RELU, bool OUTBF>
__global__ __launch_bounds__(256) void gemm_bf16_kernel(const u16* __restrict__ A,
                                                        const u16* __restrict__ Bm,
                                                        const float* __restrict__ bias,
                                                        void* __restrict__ Cout,
                                                        int M, int N, int K) {
    __shared__ __align__(16) u16 As[2 * 128 * 40];   // pitch 40 (80B, 16B aligned)
    __shared__ __align__(16) u16 Bs[2 * 32 * 136];   // pitch 136 (272B, 16B aligned)

    const int t = threadIdx.x;
    const int n0 = blockIdx.x * 128, m0 = blockIdx.y * 128;
    const int w = t >> 5, l = t & 31, ln16 = l & 15, hi = l >> 4;
    const int wr = w >> 1, wc = w & 1;               // wave grid 4 (rows) x 2 (cols)

    v8f acc[2][4] = {};

    const int ar = t >> 1, ac = (t & 1) * 16;        // A stage: 128 rows x 32 cols
    const int br = t >> 3, bc = (t & 7) * 16;        // B stage: 32 rows x 128 cols

    auto stage = [&](int k0, int buf) {
        unsigned la = lds_addr(As + buf * 128 * 40 + ar * 40 + ac);
        u64 ga = (u64)(size_t)(A + (size_t)(m0 + ar) * K + k0 + ac);
        async_b128(la, ga);
        async_b128(la + 16, ga + 16);
        unsigned lb = lds_addr(Bs + buf * 32 * 136 + br * 136 + bc);
        u64 gb = (u64)(size_t)(Bm + (size_t)(k0 + br) * N + n0 + bc);
        async_b128(lb, gb);
        async_b128(lb + 16, gb + 16);
    };

    stage(0, 0);   // prologue fill of buffer 0

    for (int k0 = 0; k0 < K; k0 += 32) {
        const int cur = (k0 >> 5) & 1;
        wait_async0();          // my fills of 'cur' have landed in LDS
        __syncthreads();        // everyone's fills landed; prev reads of cur^1 done
        if (k0 + 32 < K) stage(k0 + 32, cur ^ 1);

        const u16* as = As + cur * 128 * 40;
        const u16* bs = Bs + cur * 32 * 136;
        FragU a0, a1;
        const int mr = wr * 32 + ln16;
        a0.q[0] = *(const uint4v*)(as + mr * 40 + hi * 8);
        a0.q[1] = *(const uint4v*)(as + mr * 40 + 16 + hi * 8);
        a1.q[0] = *(const uint4v*)(as + (mr + 16) * 40 + hi * 8);
        a1.q[1] = *(const uint4v*)(as + (mr + 16) * 40 + 16 + hi * 8);
#pragma unroll
        for (int j = 0; j < 4; ++j) {
            FragU bfr;
            for (int s = 0; s < 16; ++s) ((u16*)&bfr)[s] = bs[(hi * 8 + (s & 7) + (s >> 3) * 16) * 136 + wc * 64 + j * 16 + ln16];
            acc[0][j] = __builtin_amdgcn_wmma_f32_16x16x32_bf16(
                false, a0.v, false, bfr.v, (short)0, acc[0][j], false, false);
            acc[1][j] = __builtin_amdgcn_wmma_f32_16x16x32_bf16(
                false, a1.v, false, bfr.v, (short)0, acc[1][j], false, false);
        }
    }

#pragma unroll
    for (int half = 0; half < 2; ++half) {
        const int rowb = m0 + wr * 32 + half * 16 + hi * 8;
#pragma unroll
        for (int j = 0; j < 4; ++j) {
            const int col = n0 + wc * 64 + j * 16 + ln16;
            const float bv = bias ? bias[col] : 0.f;
#pragma unroll
            for (int r = 0; r < 8; ++r) {
                float v = acc[half][j][r] + bv;
                if (RELU) v = fmaxf(v, 0.f);
                size_t idx = (size_t)(rowb + r) * N + col;
                if (OUTBF) ((u16*)Cout)[idx] = f2bf(v);
                else       ((float*)Cout)[idx] = v;
            }
        }
    }
}

// ---------------- flash attention, multiplicative mask ----------------
// grid: B*H*(S/64) blocks, 128 threads = 4 waves, wave -> 16 q rows, 64-key blocks.
__global__ __launch_bounds__(128) void flash_attn_kernel(const u16* __restrict__ Q,
                                                         const u16* __restrict__ Kb,
                                                         const u16* __restrict__ Vb,
                                                         const int* __restrict__ mask,
                                                         float* __restrict__ O) {
    __shared__ __align__(16) u16 Kt_s[64 * 72];       // [dh][key], pitch 72 (144B)
    __shared__ __align__(16) u16 V_s[64 * 72];        // [key][dh]
    __shared__ __align__(16) u16 P_s[4 * 16 * 72];    // per-wave P tile [16][64]

    const int t = threadIdx.x;
    const int w = t >> 5, l = t & 31, ln16 = l & 15, hi = l >> 4;
    const int nqb = SS / 64;
    const int qb = blockIdx.x % nqb;
    const int bh = blockIdx.x / nqb;
    const int b = bh / HH, h = bh % HH;
    const int q0 = qb * 64;

    // Q fragments (A-operand), rows = q, k-dim = dh (64 -> 2 steps)
    const int qrow = q0 + w * 16 + ln16;
    const u16* qp = Q + ((size_t)(b * SS + qrow)) * DD + h * 64;
    FragU Qf0, Qf1;
    Qf0.q[0] = *(const uint4v*)(qp + hi * 8);
    Qf0.q[1] = *(const uint4v*)(qp + 16 + hi * 8);
    Qf1.q[0] = *(const uint4v*)(qp + 32 + hi * 8);
    Qf1.q[1] = *(const uint4v*)(qp + 48 + hi * 8);

    float mx[8], sm[8];
    v8f Oc[4] = {};
#pragma unroll
    for (int r = 0; r < 8; ++r) { mx[r] = -1e30f; sm[r] = 0.f; }

    const int rn = t >> 1, rc = t & 1;    // staging: key row, dh half

    for (int kb0 = 0; kb0 < SS; kb0 += 64) {
        __syncthreads();                  // prior iteration reads of K/V done
        {
            // V block row-major: async global->LDS, 4x16B per thread
            unsigned lv = lds_addr(V_s + rn * 72 + rc * 32);
            u64 gv = (u64)(size_t)(Vb + ((size_t)(b * SS + kb0 + rn)) * DD + h * 64 + rc * 32);
            async_b128(lv, gv);
            async_b128(lv + 16, gv + 16);
            async_b128(lv + 32, gv + 32);
            async_b128(lv + 48, gv + 48);
            // K block transposed -> Kt[dh][key] via VGPR scatter
            const uint4v* ksrc = (const uint4v*)(Kb + ((size_t)(b * SS + kb0 + rn)) * DD + h * 64 + rc * 32);
            uint4v rr[4];
            rr[0] = ksrc[0]; rr[1] = ksrc[1]; rr[2] = ksrc[2]; rr[3] = ksrc[3];
            const u16* ru = (const u16*)rr;
#pragma unroll
            for (int i = 0; i < 32; ++i) Kt_s[(rc * 32 + i) * 72 + rn] = ru[i];
        }
        wait_async0();
        __syncthreads();

        // scores: 16q x 64k = 4 wmma tiles, k-dim = dh
        float st[4][8];
#pragma unroll
        for (int nt = 0; nt < 4; ++nt) {
            v8f acc = {};
            FragU bf0, bf1;
            for (int s = 0; s < 16; ++s) {
                const int kk = hi * 8 + (s & 7) + (s >> 3) * 16;
                ((u16*)&bf0)[s] = Kt_s[kk * 72 + nt * 16 + ln16];
                ((u16*)&bf1)[s] = Kt_s[(32 + kk) * 72 + nt * 16 + ln16];
            }
            acc = __builtin_amdgcn_wmma_f32_16x16x32_bf16(false, Qf0.v, false, bf0.v, (short)0, acc, false, false);
            acc = __builtin_amdgcn_wmma_f32_16x16x32_bf16(false, Qf1.v, false, bf1.v, (short)0, acc, false, false);
#pragma unroll
            for (int r = 0; r < 8; ++r) {
                const int qr = q0 + w * 16 + hi * 8 + r;
                const int kc = kb0 + nt * 16 + ln16;
                const float mf = (float)mask[(size_t)qr * SS + kc];
                st[nt][r] = acc[r] * 0.125f * mf;    // scale = 1/sqrt(64); mask multiplies score
            }
        }

        // online softmax stats (row = M, spread across 16 lanes of a half)
        float corr[8];
#pragma unroll
        for (int r = 0; r < 8; ++r) {
            float bm = fmaxf(fmaxf(st[0][r], st[1][r]), fmaxf(st[2][r], st[3][r]));
#pragma unroll
            for (int o = 8; o > 0; o >>= 1) bm = fmaxf(bm, __shfl_xor(bm, o, 16));
            const float mn = fmaxf(mx[r], bm);
            corr[r] = __expf(mx[r] - mn);
            mx[r] = mn;
            float ps = 0.f;
#pragma unroll
            for (int nt = 0; nt < 4; ++nt) { float p = __expf(st[nt][r] - mn); st[nt][r] = p; ps += p; }
#pragma unroll
            for (int o = 8; o > 0; o >>= 1) ps += __shfl_xor(ps, o, 16);
            sm[r] = sm[r] * corr[r] + ps;
        }

        // C-layout -> A-layout re-shape through per-wave LDS tile
        u16* pw = P_s + w * 16 * 72;
#pragma unroll
        for (int nt = 0; nt < 4; ++nt)
#pragma unroll
            for (int r = 0; r < 8; ++r)
                pw[(hi * 8 + r) * 72 + nt * 16 + ln16] = f2bf(st[nt][r]);

        FragU Pf0, Pf1;
        const u16* pr = P_s + w * 16 * 72 + ln16 * 72;
        Pf0.q[0] = *(const uint4v*)(pr + hi * 8);
        Pf0.q[1] = *(const uint4v*)(pr + 16 + hi * 8);
        Pf1.q[0] = *(const uint4v*)(pr + 32 + hi * 8);
        Pf1.q[1] = *(const uint4v*)(pr + 48 + hi * 8);

        // O = O*corr + P @ V   (k-dim = 64 keys -> 2 steps)
#pragma unroll
        for (int dt = 0; dt < 4; ++dt) {
#pragma unroll
            for (int r = 0; r < 8; ++r) Oc[dt][r] *= corr[r];
            FragU vf0, vf1;
            for (int s = 0; s < 16; ++s) {
                const int kk = hi * 8 + (s & 7) + (s >> 3) * 16;
                ((u16*)&vf0)[s] = V_s[kk * 72 + dt * 16 + ln16];
                ((u16*)&vf1)[s] = V_s[(32 + kk) * 72 + dt * 16 + ln16];
            }
            Oc[dt] = __builtin_amdgcn_wmma_f32_16x16x32_bf16(false, Pf0.v, false, vf0.v, (short)0, Oc[dt], false, false);
            Oc[dt] = __builtin_amdgcn_wmma_f32_16x16x32_bf16(false, Pf1.v, false, vf1.v, (short)0, Oc[dt], false, false);
        }
    }

#pragma unroll
    for (int dt = 0; dt < 4; ++dt)
#pragma unroll
        for (int r = 0; r < 8; ++r) {
            const int qr = q0 + w * 16 + hi * 8 + r;
            const int dc = h * 64 + dt * 16 + ln16;
            O[((size_t)(b * SS + qr)) * DD + dc] = Oc[dt][r] / sm[r];
        }
}

// ---------------- residual + layernorm (fp32 + bf16 outputs) ----------------
__global__ __launch_bounds__(256) void add_ln_kernel(const float* __restrict__ A,
                                                     const float* __restrict__ Bv,
                                                     const float* __restrict__ g,
                                                     const float* __restrict__ be,
                                                     float* __restrict__ Of,
                                                     u16* __restrict__ Ob) {
    const int tok = blockIdx.x;
    const int t = threadIdx.x;
    float4v x = ((const float4v*)(A + (size_t)tok * DD))[t];
    float4v y = ((const float4v*)(Bv + (size_t)tok * DD))[t];
    const float v0 = x.x + y.x, v1 = x.y + y.y, v2 = x.z + y.z, v3 = x.w + y.w;
    float s = v0 + v1 + v2 + v3;
    float ss = v0 * v0 + v1 * v1 + v2 * v2 + v3 * v3;
#pragma unroll
    for (int o = 16; o > 0; o >>= 1) { s += __shfl_xor(s, o, 32); ss += __shfl_xor(ss, o, 32); }
    __shared__ float rs[8], rss[8];
    const int w = t >> 5, l = t & 31;
    if (l == 0) { rs[w] = s; rss[w] = ss; }
    __syncthreads();
    s = 0.f; ss = 0.f;
#pragma unroll
    for (int i = 0; i < 8; ++i) { s += rs[i]; ss += rss[i]; }
    const float mean = s * (1.f / (float)DD);
    const float var = ss * (1.f / (float)DD) - mean * mean;
    const float rstd = rsqrtf(var + 1e-6f);
    float4v gg = ((const float4v*)g)[t];
    float4v bb = ((const float4v*)be)[t];
    const float o0 = (v0 - mean) * rstd * gg.x + bb.x;
    const float o1 = (v1 - mean) * rstd * gg.y + bb.y;
    const float o2 = (v2 - mean) * rstd * gg.z + bb.z;
    const float o3 = (v3 - mean) * rstd * gg.w + bb.w;
    if (Of) {
        float4v o; o.x = o0; o.y = o1; o.z = o2; o.w = o3;
        ((float4v*)(Of + (size_t)tok * DD))[t] = o;
    }
    ushort4v ob; ob.x = f2bf(o0); ob.y = f2bf(o1); ob.z = f2bf(o2); ob.w = f2bf(o3);
    ((ushort4v*)(Ob + (size_t)tok * DD))[t] = ob;
}

// ---------------- host orchestration ----------------
extern "C" void kernel_launch(void* const* d_in, const int* in_sizes, int n_in,
                              void* d_out, int out_size, void* d_ws, size_t ws_size,
                              hipStream_t stream) {
    (void)in_sizes; (void)n_in; (void)out_size; (void)ws_size;

    const float* x_in = (const float*)d_in[0];
    const int*   mask = (const int*)d_in[1];
    const float* Wq = (const float*)d_in[2];
    const float* Wk = (const float*)d_in[3];
    const float* Wv = (const float*)d_in[4];
    const float* W1 = (const float*)d_in[5];
    const float* b1 = (const float*)d_in[6];
    const float* W2 = (const float*)d_in[7];
    const float* b2 = (const float*)d_in[8];
    const float* g1 = (const float*)d_in[9];
    const float* be1 = (const float*)d_in[10];
    const float* g2 = (const float*)d_in[11];
    const float* be2 = (const float*)d_in[12];
    float* out = (float*)d_out;

    char* ws = (char*)d_ws;
    size_t off = 0;
    auto alloc = [&](size_t bytes) -> void* {
        void* p = ws + off; off += (bytes + 255) & ~(size_t)255; return p;
    };
    const size_t NX = (size_t)BSS * DD;         // 4M
    const size_t NH = (size_t)BSS * FF;         // 16M
    u16*   xb    = (u16*)alloc(NX * 2);
    u16*   qb    = (u16*)alloc(NX * 2);
    u16*   kb    = (u16*)alloc(NX * 2);
    u16*   vb    = (u16*)alloc(NX * 2);
    float* enc   = (float*)alloc(NX * 4);
    u16*   enc1b = (u16*)alloc(NX * 2);
    u16*   hb    = (u16*)alloc(NH * 2);
    float* ffn   = (float*)alloc(NX * 4);
    float* xcur  = (float*)alloc(NX * 4);
    u16*   Wqb   = (u16*)alloc((size_t)LNUM * DD * DD * 2);
    u16*   Wkb   = (u16*)alloc((size_t)LNUM * DD * DD * 2);
    u16*   Wvb   = (u16*)alloc((size_t)LNUM * DD * DD * 2);
    u16*   W1b   = (u16*)alloc((size_t)LNUM * DD * FF * 2);
    u16*   W2b   = (u16*)alloc((size_t)LNUM * FF * DD * 2);

    auto conv = [&](const float* src, u16* dst, size_t n) {
        int n4 = (int)(n / 4);
        conv_bf16_kernel<<<(n4 + 255) / 256, 256, 0, stream>>>(src, dst, n4);
    };

    conv(Wq, Wqb, (size_t)LNUM * DD * DD);
    conv(Wk, Wkb, (size_t)LNUM * DD * DD);
    conv(Wv, Wvb, (size_t)LNUM * DD * DD);
    conv(W1, W1b, (size_t)LNUM * DD * FF);
    conv(W2, W2b, (size_t)LNUM * FF * DD);
    conv(x_in, xb, NX);

    for (int l = 0; l < LNUM; ++l) {
        const float* xf = (l == 0) ? x_in : xcur;

        dim3 gQKV(DD / 128, BSS / 128);
        gemm_bf16_kernel<false, true><<<gQKV, 256, 0, stream>>>(
            xb, Wqb + (size_t)l * DD * DD, nullptr, qb, BSS, DD, DD);
        gemm_bf16_kernel<false, true><<<gQKV, 256, 0, stream>>>(
            xb, Wkb + (size_t)l * DD * DD, nullptr, kb, BSS, DD, DD);
        gemm_bf16_kernel<false, true><<<gQKV, 256, 0, stream>>>(
            xb, Wvb + (size_t)l * DD * DD, nullptr, vb, BSS, DD, DD);

        flash_attn_kernel<<<BB * HH * (SS / 64), 128, 0, stream>>>(qb, kb, vb, mask, enc);

        // enc1 = LN(x + enc) -> bf16 only (fp32 side unused)
        add_ln_kernel<<<BSS, 256, 0, stream>>>(xf, enc, g1 + (size_t)l * DD,
                                               be1 + (size_t)l * DD, nullptr, enc1b);

        dim3 gF1(FF / 128, BSS / 128);
        gemm_bf16_kernel<true, true><<<gF1, 256, 0, stream>>>(
            enc1b, W1b + (size_t)l * DD * FF, b1 + (size_t)l * FF, hb, BSS, FF, DD);

        dim3 gF2(DD / 128, BSS / 128);
        gemm_bf16_kernel<false, false><<<gF2, 256, 0, stream>>>(
            hb, W2b + (size_t)l * FF * DD, b2 + (size_t)l * DD, ffn, BSS, DD, FF);

        // x_next = LN(enc + ffn)   (residual from enc, per reference)
        float* odst = (l == LNUM - 1) ? out : xcur;
        add_ln_kernel<<<BSS, 256, 0, stream>>>(enc, ffn, g2 + (size_t)l * DD,
                                               be2 + (size_t)l * DD, odst, xb);
    }
}
